// vq_vae_17678085390782
// MI455X (gfx1250) — compile-verified, matrix operand order checked by reading, not run on hardware
//
#include <hip/hip_runtime.h>
#include <hip/hip_bf16.h>

// ---------------------------------------------------------------------------
// VQ-VAE codebook step for gfx1250 (MI455X).
// Core: fused (2*x@E^T - ||e||^2) argmax using v_wmma_f32_16x16x32_bf16 with
// async global->LDS staging of the codebook tiles (ASYNCcnt pipeline).
// ---------------------------------------------------------------------------

typedef __bf16 bf16_t;
typedef bf16_t v16bf __attribute__((ext_vector_type(16)));
typedef float  v8f   __attribute__((ext_vector_type(8)));

#define DIMK   256       // dim per group
#define NCODES 1024
#define NROWS  131072    // N * n_groups
#define NVEC   65536
#define DLAT   512
#define DECAYF 0.995f
#define OMDF   0.005f
#define EPSV   1e-4f
#define THRESH 0.99f

// output offsets (float elements, concatenated tuple order)
#define Z_OFF      0ull
#define DIFF_OFF   33554432ull
#define CODES_OFF  33554433ull
#define EMB_OFF    33685505ull
#define CS_OFF     33947649ull
#define CSUM_OFF   33948673ull
#define AVGU_OFF   34210817ull
#define USAGE_OFF  34210818ull
#define ENT_OFF    34210819ull

// workspace byte offsets
#define WS_EBF   0ull        // NCODES*DIMK bf16 (512 KB)
#define WS_E2    524288ull   // NCODES f32
#define WS_IDX   528384ull   // NROWS i32
#define WS_CNT   1052672ull  // NCODES f32
#define WS_SUM   1056768ull  // NCODES*DIMK f32 (1 MB)
#define WS_SCAL  2105344ull  // [0]=diff_acc, [1]=n

// LDS staging: 16 code rows x 512B, padded to 528B pitch (bank spread)
#define BPITCH 528
#define BBUF   (16 * BPITCH)   // 8448 B per buffer

// one 16-byte async global->LDS copy (GVS mode, tracked by ASYNCcnt)
__device__ __forceinline__ void async_copy16(unsigned lds_off,
                                             const void* gbase,
                                             unsigned goff) {
  asm volatile("global_load_async_to_lds_b128 %0, %1, %2"
               :: "v"(lds_off), "v"(goff), "s"(gbase)
               : "memory");
}

// --- prep: embedding f32 -> bf16 + ||e||^2 per code -------------------------
__global__ __launch_bounds__(256) void k_prep(const float* __restrict__ emb,
                                              unsigned short* __restrict__ ebf,
                                              float* __restrict__ e2) {
  __shared__ float sred[256];
  const int c = blockIdx.x, d = threadIdx.x;
  const float v = emb[c * DIMK + d];
  ebf[c * DIMK + d] = __builtin_bit_cast(unsigned short, (bf16_t)v);
  sred[d] = v * v;
  __syncthreads();
  for (int s = 128; s > 0; s >>= 1) {
    if (d < s) sred[d] += sred[d + s];
    __syncthreads();
  }
  if (d == 0) e2[c] = sred[0];
}

// --- zero scratch (counts + sums + scalars, contiguous) ---------------------
__global__ void k_zero(float* __restrict__ p, int n) {
  const int i = blockIdx.x * blockDim.x + threadIdx.x;
  if (i < n) p[i] = 0.0f;
}

// --- fused GEMM + argmax ----------------------------------------------------
// Block = 8 waves, each wave owns one 16-row tile (A resident in VGPRs,
// f32->bf16 converted in-register). B = E^T tiles (16 codes x 256 bf16 =
// 8 KB, contiguous in the bf16 copy) are async-copied into a double-buffered
// LDS stage shared by all 8 waves: cuts L2 B-traffic 8x (4.2 GB -> 525 MB).
// score = 2*dot - ||e||^2 maximized  ==  distance minimized.
__global__ __launch_bounds__(256) void k_gemm_argmax(
    const float* __restrict__ x, const unsigned short* __restrict__ ebf,
    const float* __restrict__ e2, int* __restrict__ idxout) {
  __shared__ char bsm[2 * BBUF];

  const int t     = threadIdx.x;
  const int wave  = t >> 5;
  const int lane  = t & 31;
  const int rlane = lane & 15;
  const int half  = lane >> 4;
  const int rt    = blockIdx.x * 8 + wave;   // 0..8191 row tiles

  const int row = rt * 16 + rlane;
  // Load & convert 8 resident A fragments (16x32 bf16 each; ISA 7.12.2 layout:
  // per lane two contiguous 8-element chunks at k0 and k0+16).
  v16bf afrag[8];
#pragma unroll
  for (int f = 0; f < 8; ++f) {
    const int k0 = f * 32 + half * 8;
    const float4* p0 = (const float4*)(x + (size_t)row * DIMK + k0);
    const float4* p1 = (const float4*)(x + (size_t)row * DIMK + k0 + 16);
    const float4 c00 = p0[0], c01 = p0[1];
    const float4 c10 = p1[0], c11 = p1[1];
    v16bf a;
    a[0]  = (bf16_t)c00.x; a[1]  = (bf16_t)c00.y;
    a[2]  = (bf16_t)c00.z; a[3]  = (bf16_t)c00.w;
    a[4]  = (bf16_t)c01.x; a[5]  = (bf16_t)c01.y;
    a[6]  = (bf16_t)c01.z; a[7]  = (bf16_t)c01.w;
    a[8]  = (bf16_t)c10.x; a[9]  = (bf16_t)c10.y;
    a[10] = (bf16_t)c10.z; a[11] = (bf16_t)c10.w;
    a[12] = (bf16_t)c11.x; a[13] = (bf16_t)c11.y;
    a[14] = (bf16_t)c11.z; a[15] = (bf16_t)c11.w;
    afrag[f] = a;
  }

  // Per-thread staging addresses: 512 x 16B chunks per 8KB tile, 2 per thread.
  const unsigned lds_base = (unsigned)(size_t)(&bsm[0]);
  const int c0 = t, c1 = t + 256;
  const unsigned l0 = (unsigned)((c0 >> 5) * BPITCH + (c0 & 31) * 16);
  const unsigned l1 = (unsigned)((c1 >> 5) * BPITCH + (c1 & 31) * 16);
  const unsigned g0 = (unsigned)(c0 * 16);
  const unsigned g1 = (unsigned)(c1 * 16);

  // Prologue: stage tiles 0 and 1.
  async_copy16(lds_base + 0u * BBUF + l0, ebf, 0u * 8192u + g0);
  async_copy16(lds_base + 0u * BBUF + l1, ebf, 0u * 8192u + g1);
  async_copy16(lds_base + 1u * BBUF + l0, ebf, 1u * 8192u + g0);
  async_copy16(lds_base + 1u * BBUF + l1, ebf, 1u * 8192u + g1);

  float bestv[8];
  int   besti[8];
#pragma unroll
  for (int r = 0; r < 8; ++r) { bestv[r] = -3.4e38f; besti[r] = 0; }

  for (int ct = 0; ct < NCODES / 16; ++ct) {
    // Async loads complete in order: <=2 outstanding => tile `ct` has landed.
    if (ct < 63) asm volatile("s_wait_asynccnt 0x2" ::: "memory");
    else         asm volatile("s_wait_asynccnt 0x0" ::: "memory");
    __syncthreads();

    const unsigned bufo = (ct & 1) ? (unsigned)BBUF : 0u;
    const char* bp = &bsm[bufo + (unsigned)(rlane * BPITCH + half * 16)];
    v8f acc0 = {}, acc1 = {};
#pragma unroll
    for (int f = 0; f < 8; f += 2) {
      union { uint4 q[2]; v16bf v; } u0, u1;
      u0.q[0] = *(const uint4*)(bp + f * 64);
      u0.q[1] = *(const uint4*)(bp + f * 64 + 32);
      u1.q[0] = *(const uint4*)(bp + f * 64 + 64);
      u1.q[1] = *(const uint4*)(bp + f * 64 + 96);
      acc0 = __builtin_amdgcn_wmma_f32_16x16x32_bf16(
          false, afrag[f],     false, u0.v, (short)0, acc0, false, false);
      acc1 = __builtin_amdgcn_wmma_f32_16x16x32_bf16(
          false, afrag[f + 1], false, u1.v, (short)0, acc1, false, false);
    }
    const int code = ct * 16 + rlane;   // C col = lane&15 -> E row `code`
    const float e2c = e2[code];
#pragma unroll
    for (int r = 0; r < 8; ++r) {
      const float sc = 2.0f * (acc0[r] + acc1[r]) - e2c;
      if (sc > bestv[r]) { bestv[r] = sc; besti[r] = code; }
    }
    __syncthreads();  // all waves done with buf[ct&1] before refilling it
    if (ct + 2 < NCODES / 16) {
      const unsigned gb = (unsigned)((ct + 2) * 8192);
      async_copy16(lds_base + bufo + l0, ebf, gb + g0);
      async_copy16(lds_base + bufo + l1, ebf, gb + g1);
    }
  }

  // Reduce over the 16 lanes of this half (masks 1..8 stay within the half).
#pragma unroll
  for (int r = 0; r < 8; ++r) {
    float v = bestv[r]; int i = besti[r];
#pragma unroll
    for (int m = 1; m <= 8; m <<= 1) {
      const float ov = __shfl_xor(v, m, 32);
      const int   oi = __shfl_xor(i, m, 32);
      if (ov > v || (ov == v && oi < i)) { v = ov; i = oi; }
    }
    bestv[r] = v; besti[r] = i;
  }
  if (rlane == 0) {
#pragma unroll
    for (int r = 0; r < 8; ++r)
      idxout[rt * 16 + half * 8 + r] = besti[r];  // C VGPR r -> row half*8+r
  }
}

// --- gather z, codes, diff, and segment-sum scatter (EMA numerators) --------
__global__ __launch_bounds__(256) void k_scatter(
    const float* __restrict__ x, const float* __restrict__ emb,
    const int* __restrict__ idxbuf, float* __restrict__ zout,
    float* __restrict__ codesout, float* __restrict__ sums,
    float* __restrict__ counts, float* __restrict__ diff_acc) {
  const int wave = threadIdx.x >> 5, lane = threadIdx.x & 31;
  const int r = blockIdx.x * 8 + wave;         // row of x_ (131072 total)
  const int idx = idxbuf[r];
  float ld = 0.0f;
#pragma unroll
  for (int j = 0; j < 8; ++j) {
    const int d = lane + j * 32;
    const float e  = emb[(size_t)idx * DIMK + d];
    const float xv = x[(size_t)r * DIMK + d];
    zout[(size_t)r * DIMK + d] = e;            // z == quantize (straight-through)
    const float df = e - xv;
    ld += df * df;
    atomicAdd(&sums[(size_t)idx * DIMK + d], xv);
  }
#pragma unroll
  for (int m = 16; m > 0; m >>= 1) ld += __shfl_xor(ld, m, 32);
  if (lane == 0) {
    atomicAdd(&counts[idx], 1.0f);
    atomicAdd(diff_acc, ld);
    codesout[r] = (float)idx;
  }
}

// --- per-code EMA size + scalar reductions (n, usage, entropy, diff) --------
__global__ __launch_bounds__(1024) void k_cluster(
    const float* __restrict__ cs, const float* __restrict__ counts,
    float* __restrict__ out, const float* __restrict__ diff_acc,
    float* __restrict__ ws_n) {
  __shared__ float sn[1024], su[1024], se[1024];
  const int t = threadIdx.x;
  const float cnt = counts[t];
  const float ncs = DECAYF * cs[t] + OMDF * cnt;
  out[CS_OFF + t] = ncs;
  const float used = (ncs >= THRESH) ? 1.0f : 0.0f;
  const float pr = cnt / (float)NROWS;  // counts.sum() == NROWS
  sn[t] = ncs; su[t] = used; se[t] = pr * logf(pr + 1e-5f);
  __syncthreads();
  for (int s = 512; s > 0; s >>= 1) {
    if (t < s) { sn[t] += sn[t + s]; su[t] += su[t + s]; se[t] += se[t + s]; }
    __syncthreads();
  }
  if (t == 0) {
    out[DIFF_OFF]  = diff_acc[0] / 33554432.0f;  // mean over N*DLAT
    out[AVGU_OFF]  = su[0] / (float)NCODES;
    out[USAGE_OFF] = su[0];
    out[ENT_OFF]   = -se[0];
    ws_n[0] = sn[0];
  }
}

// --- new_cluster_sum + new_embedding (centers or pseudo-random restarts) ----
__global__ __launch_bounds__(256) void k_embed(
    const float* __restrict__ cs, const float* __restrict__ counts,
    const float* __restrict__ csum, const float* __restrict__ sums,
    const float* __restrict__ x, const float* __restrict__ ws_n,
    float* __restrict__ out) {
  const int c = blockIdx.x, d = threadIdx.x;
  const float ncs = DECAYF * cs[c] + OMDF * counts[c];
  const float n = ws_n[0];
  const float cnt = (ncs + EPSV) / (n + (float)NCODES * EPSV) * n;
  const float ncsum = DECAYF * csum[(size_t)c * DIMK + d] + OMDF * sums[(size_t)c * DIMK + d];
  out[CSUM_OFF + (size_t)c * DIMK + d] = ncsum;
  const float center = ncsum / cnt;
  // JAX PRNG permutation is not reproducible on device; injective pseudo-perm.
  const unsigned perm = ((unsigned)c * 40507u + 12345u) & (NROWS - 1u);
  const float rnd = x[(size_t)perm * DIMK + d];
  const float used = (ncs >= THRESH) ? 1.0f : 0.0f;
  out[EMB_OFF + (size_t)c * DIMK + d] = used * center + (1.0f - used) * rnd;
}

extern "C" void kernel_launch(void* const* d_in, const int* in_sizes, int n_in,
                              void* d_out, int out_size, void* d_ws, size_t ws_size,
                              hipStream_t stream) {
  const float* x    = (const float*)d_in[0];   // [65536, 512] -> rows of 256
  const float* emb  = (const float*)d_in[1];   // [1024, 256]
  const float* cs   = (const float*)d_in[2];   // [1024]
  const float* csum = (const float*)d_in[3];   // [1024, 256]
  float* out = (float*)d_out;
  char*  ws  = (char*)d_ws;

  unsigned short* ebf  = (unsigned short*)(ws + WS_EBF);
  float* e2   = (float*)(ws + WS_E2);
  int*   idx  = (int*)(ws + WS_IDX);
  float* cnts = (float*)(ws + WS_CNT);
  float* sums = (float*)(ws + WS_SUM);
  float* scal = (float*)(ws + WS_SCAL);

  k_prep<<<NCODES, 256, 0, stream>>>(emb, ebf, e2);
  // counts + sums + scalars are contiguous: zero them every call
  const int nzero = NCODES + NCODES * DIMK + 8;
  k_zero<<<(nzero + 255) / 256, 256, 0, stream>>>(cnts, nzero);
  k_gemm_argmax<<<NROWS / 16 / 8, 256, 0, stream>>>(x, ebf, e2, idx);
  k_scatter<<<NROWS / 8, 256, 0, stream>>>(x, emb, idx, out + Z_OFF,
                                           out + CODES_OFF, sums, cnts, scal);
  k_cluster<<<1, 1024, 0, stream>>>(cs, cnts, out, scal, scal + 1);
  k_embed<<<NCODES, 256, 0, stream>>>(cs, cnts, csum, sums, x, scal + 1, out);
}
